// Linear3Bit_40759239639814
// MI455X (gfx1250) — compile-verified, matrix operand order checked by reading, not run on hardware
//
#include <hip/hip_runtime.h>
#include <hip/hip_fp16.h>

typedef __attribute__((ext_vector_type(16))) _Float16 v16h;
typedef __attribute__((ext_vector_type(8)))  float    v8f;

#define IN_F   4096
#define OUT_F  4096
#define GSIZE  32
#define GPACK  12
#define GPR    (IN_F / GSIZE)    // 128 groups per weight row
#define NGROUPS (OUT_F * GPR)    // 524288

#define BM 128
#define BN 128

// Fused-path tile depth
#define BKF 64
#define ASTRIDEF (BKF + 4)       // 68 halves

// f16-GEMM-path tile depth
#define BK 128
#define ASTRIDE (BK + 8)         // 136 halves -> 272B row stride (16B aligned, bank-skewed)

// Extract 3-bit field i (0..31) from a 96-bit little-endian bitstream
// (lo = dwords[1]:dwords[0], d2 = dwords[2]); matches reference packing.
__device__ __forceinline__ uint32_t q3_extract(uint64_t lo, uint32_t d2, int i) {
    if (i < 21)  return (uint32_t)(lo >> (3 * i)) & 7u;
    if (i == 21) return (uint32_t)(((lo >> 63) | ((uint64_t)d2 << 1))) & 7u;
    return (d2 >> (3 * i - 64)) & 7u;
}

// ---------------------------------------------------------------------------
// Kernel 1: dequantize all of W once -> f16 [OUT_F][IN_F] (one group/thread)
// ---------------------------------------------------------------------------
__global__ __launch_bounds__(256)
void q3_dequant_w_kernel(const unsigned char* __restrict__ wq,
                         const _Float16* __restrict__ wnorm,
                         _Float16* __restrict__ Wh)
{
    const int g = blockIdx.x * 256 + threadIdx.x;       // group index
    const uint32_t* pw = (const uint32_t*)(wq + (size_t)g * GPACK);
    const uint32_t d0 = pw[0], d1 = pw[1], d2 = pw[2];
    const uint64_t lo = ((uint64_t)d1 << 32) | d0;
    const float s = (float)wnorm[g] * (1.0f / 7.0f);

    union { _Float16 h[32]; uint4 q[4]; } o;
#pragma unroll
    for (int i = 0; i < 32; ++i)
        o.h[i] = (_Float16)(s * (float)(2 * (int)q3_extract(lo, d2, i) - 7));

    uint4* dst = (uint4*)(Wh + (size_t)g * GSIZE);       // 64B per group, 16B aligned
#pragma unroll
    for (int j = 0; j < 4; ++j) dst[j] = o.q[j];
}

// ---------------------------------------------------------------------------
// Kernel 2: convert x (f32) -> f16 once (8 elements/thread)
// ---------------------------------------------------------------------------
__global__ __launch_bounds__(256)
void x_to_f16_kernel(const float* __restrict__ x, _Float16* __restrict__ Xh, int n8)
{
    const int i = blockIdx.x * 256 + threadIdx.x;
    if (i >= n8) return;
    const float4 a = ((const float4*)x)[2 * i];
    const float4 b = ((const float4*)x)[2 * i + 1];
    union { _Float16 h[8]; uint4 u; } p;
    p.h[0] = (_Float16)a.x; p.h[1] = (_Float16)a.y;
    p.h[2] = (_Float16)a.z; p.h[3] = (_Float16)a.w;
    p.h[4] = (_Float16)b.x; p.h[5] = (_Float16)b.y;
    p.h[6] = (_Float16)b.z; p.h[7] = (_Float16)b.w;
    ((uint4*)Xh)[i] = p.u;
}

// ---------------------------------------------------------------------------
// Kernel 3: pure f16 WMMA GEMM  out = Xh * Wh^T + bias
// 128x128 block tile, BK=128, 8 waves, 64x32 per wave (4x2 WMMA tiles)
// ---------------------------------------------------------------------------
__global__ __launch_bounds__(256, 2)
void gemm_f16_wmma_kernel(const _Float16* __restrict__ Xh,
                          const _Float16* __restrict__ Wh,
                          const _Float16* __restrict__ bias,
                          float* __restrict__ out)
{
    __shared__ _Float16 ldsA[BM * ASTRIDE];
    __shared__ _Float16 ldsB[BN * ASTRIDE];

    const int t    = threadIdx.x;
    const int lane = t & 31;
    const int wave = t >> 5;
    const int wm   = wave >> 2;
    const int wn   = wave & 3;
    const int lr   = lane & 15;
    const int kg   = lane >> 4;

    const int N0 = blockIdx.x * BN;
    const int M0 = blockIdx.y * BM;

    v8f acc[4][2];
    const v8f vzero = {};
#pragma unroll
    for (int ms = 0; ms < 4; ++ms)
#pragma unroll
        for (int ns = 0; ns < 2; ++ns)
            acc[ms][ns] = vzero;

    union Frag { v16h v; uint32_t u[8]; };

    for (int k0 = 0; k0 < IN_F; k0 += BK) {
        // Stage A and B tiles: 2048 uint4 each, 8 per thread per matrix.
#pragma unroll
        for (int it = 0; it < 8; ++it) {
            int idx = it * 256 + t;
            int row = idx >> 4;                 // 16 uint4 (128 halves) per row
            int c8  = (idx & 15) << 3;          // half offset within row
            *(uint4*)&ldsA[row * ASTRIDE + c8] =
                *(const uint4*)(Xh + (size_t)(M0 + row) * IN_F + k0 + c8);
            *(uint4*)&ldsB[row * ASTRIDE + c8] =
                *(const uint4*)(Wh + (size_t)(N0 + row) * IN_F + k0 + c8);
        }

        // Prefetch next K-tile (global_prefetch_b8); one line per thread.
        if (k0 + BK < IN_F) {
            __builtin_prefetch(Xh + (size_t)(M0 + (t >> 1)) * IN_F + (k0 + BK) + ((t & 1) << 6), 0, 1);
            __builtin_prefetch(Wh + (size_t)(N0 + (t >> 1)) * IN_F + (k0 + BK) + ((t & 1) << 6), 0, 1);
        }

        __syncthreads();

#pragma unroll
        for (int ks = 0; ks < BK; ks += 32) {
            Frag a[4], b[2];
#pragma unroll
            for (int ms = 0; ms < 4; ++ms) {
                const int arow = wm * 64 + ms * 16 + lr;
#pragma unroll
                for (int v = 0; v < 8; ++v) {
                    int k = ks + ((v & 3) << 1) + ((v >> 2) << 4) + (kg << 3);
                    a[ms].u[v] = *(const uint32_t*)&ldsA[arow * ASTRIDE + k];
                }
            }
#pragma unroll
            for (int ns = 0; ns < 2; ++ns) {
                const int brow = wn * 32 + ns * 16 + lr;
#pragma unroll
                for (int v = 0; v < 8; ++v) {
                    int k = ks + ((v & 3) << 1) + ((v >> 2) << 4) + (kg << 3);
                    b[ns].u[v] = *(const uint32_t*)&ldsB[brow * ASTRIDE + k];
                }
            }
#pragma unroll
            for (int ms = 0; ms < 4; ++ms)
#pragma unroll
                for (int ns = 0; ns < 2; ++ns)
                    acc[ms][ns] = __builtin_amdgcn_wmma_f32_16x16x32_f16(
                        false, a[ms].v, false, b[ns].v,
                        (short)0, acc[ms][ns], false, false);
        }

        __syncthreads();
    }

#pragma unroll
    for (int ms = 0; ms < 4; ++ms)
#pragma unroll
        for (int ns = 0; ns < 2; ++ns) {
            const int col = N0 + wn * 32 + ns * 16 + lr;
            const float bv = (float)bias[col];
#pragma unroll
            for (int v = 0; v < 8; ++v) {
                const int row = M0 + wm * 64 + ms * 16 + v + (kg << 3);
                out[(size_t)row * OUT_F + col] = acc[ms][ns][v] + bv;
            }
        }
}

// ---------------------------------------------------------------------------
// Fallback: fused dequant + GEMM (used only if d_ws is too small)
// ---------------------------------------------------------------------------
__global__ __launch_bounds__(256, 2)
void q3_gemm_fused_kernel(const float* __restrict__ x,
                          const unsigned char* __restrict__ wq,
                          const _Float16* __restrict__ wnorm,
                          const _Float16* __restrict__ bias,
                          float* __restrict__ out)
{
    __shared__ _Float16 ldsA[BM * ASTRIDEF];
    __shared__ _Float16 ldsB[BN * ASTRIDEF];

    const int t    = threadIdx.x;
    const int lane = t & 31;
    const int wave = t >> 5;
    const int wm   = wave >> 2;
    const int wn   = wave & 3;
    const int lr   = lane & 15;
    const int kg   = lane >> 4;

    const int N0 = blockIdx.x * BN;
    const int M0 = blockIdx.y * BM;

    v8f acc[4][2];
    const v8f vzero = {};
#pragma unroll
    for (int ms = 0; ms < 4; ++ms)
#pragma unroll
        for (int ns = 0; ns < 2; ++ns)
            acc[ms][ns] = vzero;

    const int    bn    = t >> 1;
    const int    kh    = t & 1;
    const size_t gbase = (size_t)(N0 + bn) * GPR + kh;

    union Frag { v16h v; uint32_t u[8]; };

    for (int k0 = 0; k0 < IN_F; k0 += BKF) {
#pragma unroll
        for (int it = 0; it < 8; ++it) {
            int idx = it * 256 + t;
            int row = idx >> 4;
            int c4  = (idx & 15) << 2;
            const float4 v = *(const float4*)(x + (size_t)(M0 + row) * IN_F + k0 + c4);
            union { _Float16 h[4]; uint2 u; } p;
            p.h[0] = (_Float16)v.x; p.h[1] = (_Float16)v.y;
            p.h[2] = (_Float16)v.z; p.h[3] = (_Float16)v.w;
            *(uint2*)&ldsA[row * ASTRIDEF + c4] = p.u;
        }
        {
            const size_t g = gbase + (size_t)(k0 >> 5);
            const uint32_t* pw = (const uint32_t*)(wq + g * GPACK);
            uint32_t d0 = pw[0], d1 = pw[1], d2 = pw[2];
            uint64_t lo = ((uint64_t)d1 << 32) | d0;
            const float s = (float)wnorm[g] * (1.0f / 7.0f);
            uint32_t* dst32 = (uint32_t*)&ldsB[bn * ASTRIDEF + (kh << 5)];
#pragma unroll
            for (int i = 0; i < 32; i += 2) {
                float w0 = s * (float)(2 * (int)q3_extract(lo, d2, i)     - 7);
                float w1 = s * (float)(2 * (int)q3_extract(lo, d2, i + 1) - 7);
                union { _Float16 h[2]; uint32_t u; } pk;
                pk.h[0] = (_Float16)w0; pk.h[1] = (_Float16)w1;
                dst32[i >> 1] = pk.u;
            }
        }
        __syncthreads();

#pragma unroll
        for (int ks = 0; ks < BKF; ks += 32) {
            Frag a[4], b[2];
#pragma unroll
            for (int ms = 0; ms < 4; ++ms) {
                const int arow = wm * 64 + ms * 16 + lr;
#pragma unroll
                for (int v = 0; v < 8; ++v) {
                    int k = ks + ((v & 3) << 1) + ((v >> 2) << 4) + (kg << 3);
                    a[ms].u[v] = *(const uint32_t*)&ldsA[arow * ASTRIDEF + k];
                }
            }
#pragma unroll
            for (int ns = 0; ns < 2; ++ns) {
                const int brow = wn * 32 + ns * 16 + lr;
#pragma unroll
                for (int v = 0; v < 8; ++v) {
                    int k = ks + ((v & 3) << 1) + ((v >> 2) << 4) + (kg << 3);
                    b[ns].u[v] = *(const uint32_t*)&ldsB[brow * ASTRIDEF + k];
                }
            }
#pragma unroll
            for (int ms = 0; ms < 4; ++ms)
#pragma unroll
                for (int ns = 0; ns < 2; ++ns)
                    acc[ms][ns] = __builtin_amdgcn_wmma_f32_16x16x32_f16(
                        false, a[ms].v, false, b[ns].v,
                        (short)0, acc[ms][ns], false, false);
        }
        __syncthreads();
    }

#pragma unroll
    for (int ms = 0; ms < 4; ++ms)
#pragma unroll
        for (int ns = 0; ns < 2; ++ns) {
            const int col = N0 + wn * 32 + ns * 16 + lr;
            const float bv = (float)bias[col];
#pragma unroll
            for (int v = 0; v < 8; ++v) {
                const int row = M0 + wm * 64 + ms * 16 + v + (kg << 3);
                out[(size_t)row * OUT_F + col] = acc[ms][ns][v] + bv;
            }
        }
}

// ---------------------------------------------------------------------------
extern "C" void kernel_launch(void* const* d_in, const int* in_sizes, int n_in,
                              void* d_out, int out_size, void* d_ws, size_t ws_size,
                              hipStream_t stream) {
    const float*         x     = (const float*)d_in[0];
    const unsigned char* wq    = (const unsigned char*)d_in[1];
    const _Float16*      wnorm = (const _Float16*)d_in[2];
    const _Float16*      bias  = (const _Float16*)d_in[3];
    float*               out   = (float*)d_out;

    const int    M     = in_sizes[0] / IN_F;                 // 8192
    const size_t needW = (size_t)OUT_F * IN_F * sizeof(_Float16);      // 32 MB
    const size_t needX = (size_t)in_sizes[0] * sizeof(_Float16);       // 64 MB

    if (ws_size >= needW + needX) {
        _Float16* Wh = (_Float16*)d_ws;
        _Float16* Xh = (_Float16*)((char*)d_ws + needW);

        q3_dequant_w_kernel<<<NGROUPS / 256, 256, 0, stream>>>(wq, wnorm, Wh);

        const int n8 = in_sizes[0] / 8;
        x_to_f16_kernel<<<(n8 + 255) / 256, 256, 0, stream>>>(x, Xh, n8);

        dim3 grid(OUT_F / BN, M / BM);
        gemm_f16_wmma_kernel<<<grid, 256, 0, stream>>>(Xh, Wh, bias, out);
    } else {
        dim3 grid(OUT_F / BN, M / BM);
        q3_gemm_fused_kernel<<<grid, 256, 0, stream>>>(x, wq, wnorm, bias, out);
    }
}
